// HeteroGNN_68504728371752
// MI455X (gfx1250) — matrix operand order repaired and checked by reading, not compiled, not run on hardware
//
#include <hip/hip_runtime.h>

static const int kNPROP = 20000;
static const int kNBR   = 50000;
static const int kEPP   = 150000;
static const int kEBP   = 200000;
static const int kEBB   = 300000;
static const int kDN = 64;
static const int kDE = 32;
static const int kH  = 128;

static inline int cdiv(long a, long b) { return (int)((a + b - 1) / b); }

typedef __attribute__((ext_vector_type(16))) __bf16          bf16x16;
typedef __attribute__((ext_vector_type(8)))  float           f32x8;
typedef __attribute__((ext_vector_type(8)))  unsigned short  ushort8;

__device__ __forceinline__ unsigned short f2bfu(float f) {
  unsigned u = __float_as_uint(f);
  unsigned r = (u + 0x7FFFu + ((u >> 16) & 1u)) >> 16;  // RNE
  return (unsigned short)r;
}

__device__ __forceinline__ void atomicMaxF(float* addr, float v) {
  if (v >= 0.0f) atomicMax((int*)addr, __float_as_int(v));
  else           atomicMin((unsigned int*)addr, __float_as_uint(v));
}

// ---------------------------------------------------------------------------
// Y[M,128] = act(X[M,K] @ W[K,128] + bias); one 16-row stripe per block,
// wave w owns the 16x16 tile at columns [16w, 16w+16). bf16 WMMA, f32 acc.
// K is a compile-time power of two (64 or 128); M % 16 == 0 at every call.
// LDS pitch K+8 halfwords: keeps every fragment pointer 16B-aligned (272=16*17)
// while spreading row bases 4 banks apart (pitch 256B would alias to bank 0).
// Each WMMA operand pair is fed by 4x ds_load_b128 (no scalar u16 gathers).
// ---------------------------------------------------------------------------
template <int K>
__global__ void __launch_bounds__(256)
k_gemm_wmma(const float* __restrict__ X, const float* __restrict__ W,
            const float* __restrict__ bias, float* __restrict__ Y, float slope) {
  constexpr int KP = K + 8;                 // padded pitch in halfwords
  constexpr int KSH = (K == 128) ? 7 : 6;   // log2(K)
  __shared__ unsigned short sA[16 * KP];    // A tile, row-major [16][KP]
  __shared__ unsigned short sBT[128 * KP];  // W transposed    [col][KP]
  const int tid = threadIdx.x;
  const long m0 = (long)blockIdx.x * 16;

  // stage A (16 x K) as bf16
  for (int i = tid; i < 16 * K; i += 256) {
    int r = i >> KSH, c = i & (K - 1);
    sA[r * KP + c] = f2bfu(X[m0 * K + i]);
  }
  // stage W transposed: sBT[c][k] = bf16(W[k*128 + c]); coalesced global reads
  for (int i = tid; i < K * 128; i += 256)
    sBT[(i & 127) * KP + (i >> 7)] = f2bfu(W[i]);
  __syncthreads();

  const int lane = tid & 31;
  const int col0 = (tid >> 5) * 16;
  const int r  = lane & 15;
  const int hi = lane >> 4;  // lane half selects K sub-block per ISA layout

  const unsigned short* aBase = sA + r * KP + hi * 8;
  const unsigned short* bBase = sBT + (col0 + r) * KP + hi * 8;

  f32x8 acc = {};
#pragma unroll
  for (int kb = 0; kb < K; kb += 32) {
    // A lane data: K = kb+hi*8 .. +7  and  K = kb+16+hi*8 .. +7 (contiguous)
    ushort8 a0 = *(const ushort8*)(aBase + kb);
    ushort8 a1 = *(const ushort8*)(aBase + kb + 16);
    // B lane data: K = kb+hi*8 .. +7  and  K = kb+16+hi*8 .. +7 at fixed column
    ushort8 b0 = *(const ushort8*)(bBase + kb);
    ushort8 b1 = *(const ushort8*)(bBase + kb + 16);
    bf16x16 a = __builtin_bit_cast(bf16x16,
        __builtin_shufflevector(a0, a1, 0, 1, 2, 3, 4, 5, 6, 7,
                                        8, 9, 10, 11, 12, 13, 14, 15));
    bf16x16 b = __builtin_bit_cast(bf16x16,
        __builtin_shufflevector(b0, b1, 0, 1, 2, 3, 4, 5, 6, 7,
                                        8, 9, 10, 11, 12, 13, 14, 15));
    acc = __builtin_amdgcn_wmma_f32_16x16x32_bf16(false, a, false, b,
                                                  (short)0, acc, false, false);
  }

  const int n = col0 + r;
  const long rb = m0 + hi * 8;
  const float bn = bias ? bias[n] : 0.0f;
#pragma unroll
  for (int i = 0; i < 8; ++i) {
    float v = acc[i] + bn;
    v = v > 0.0f ? v : v * slope;  // slope==1 -> identity
    Y[(rb + i) * 128 + n] = v;
  }
}

// out[r] = X[r,:128] . v  (wave per row)
__global__ void k_rowdot(const float* __restrict__ X, const float* __restrict__ v,
                         float* __restrict__ out, int M) {
  int gid = blockIdx.x * blockDim.x + threadIdx.x;
  int row = gid >> 5, lane = gid & 31;
  if (row >= M) return;
  float4 a = ((const float4*)(X + (size_t)row * kH))[lane];
  float4 b = ((const float4*)v)[lane];
  float s = a.x * b.x + a.y * b.y + a.z * b.z + a.w * b.w;
  for (int o = 16; o > 0; o >>= 1) s += __shfl_down(s, o);
  if (lane == 0) out[row] = s;
}

// out[i] = sum_j M[i,j]*a[j], 128x128; one block of 128 threads
__global__ void k_matvec128(const float* __restrict__ Mw, const float* __restrict__ a,
                            float* __restrict__ out) {
  __shared__ float sa[128];
  int t = threadIdx.x;
  sa[t] = a[t];
  __syncthreads();
  float s = 0.0f;
  for (int j = 0; j < 128; ++j) s += Mw[(size_t)t * 128 + j] * sa[j];
  out[t] = s;
}

// out[0] = a . b (128-dim), one wave
__global__ void k_dot128(const float* __restrict__ a, const float* __restrict__ b,
                         float* __restrict__ out) {
  int lane = threadIdx.x;
  float s = 0.0f;
  for (int i = lane; i < 128; i += 32) s += a[i] * b[i];
  for (int o = 16; o > 0; o >>= 1) s += __shfl_down(s, o);
  if (lane == 0) out[0] = s;
}

__global__ void k_fill(float* __restrict__ p, float v, int n) {
  int i = blockIdx.x * blockDim.x + threadIdx.x;
  if (i < n) p[i] = v;
}

// out[idx] = b1[idx&127] + (b2 ? b2[idx&127] : 0)
__global__ void k_init2(float* __restrict__ out, const float* __restrict__ b1,
                        const float* __restrict__ b2, int n) {
  int i = blockIdx.x * blockDim.x + threadIdx.x;
  if (i >= n) return;
  int f = i & (kH - 1);
  out[i] = b1[f] + (b2 ? b2[f] : 0.0f);
}

// Fused: p = lrelu(ea[e,:De] @ We + be)  (128-dim, transient in regs),
// a_edge[e] = p . v ; optionally accumulate mean(p)*invE into meanAcc.
// One wave per edge; lane owns 4 output features.
__global__ void __launch_bounds__(256)
k_edge_alpha(const float* __restrict__ ea, int De,
             const float* __restrict__ We, const float* __restrict__ be,
             const float* __restrict__ v, float* __restrict__ a_edge, int E,
             float* __restrict__ meanAcc, float invE) {
  __shared__ float smean[128];
  const bool doMean = (meanAcc != nullptr);
  const int tid = threadIdx.x;
  if (doMean) {
    for (int i = tid; i < 128; i += blockDim.x) smean[i] = 0.0f;
    __syncthreads();
  }
  const int e = (blockIdx.x * blockDim.x + tid) >> 5;
  const int lane = tid & 31;
  if (e < E) {
    const int f0 = lane * 4;
    float p0 = be[f0], p1 = be[f0 + 1], p2 = be[f0 + 2], p3 = be[f0 + 3];
    const float* row = ea + (size_t)e * De;
    for (int k = 0; k < De; ++k) {
      float x = row[k];
      const float* wr = We + (size_t)k * 128 + f0;
      p0 += x * wr[0]; p1 += x * wr[1]; p2 += x * wr[2]; p3 += x * wr[3];
    }
    p0 = p0 > 0.f ? p0 : 0.01f * p0;
    p1 = p1 > 0.f ? p1 : 0.01f * p1;
    p2 = p2 > 0.f ? p2 : 0.01f * p2;
    p3 = p3 > 0.f ? p3 : 0.01f * p3;
    float s = p0 * v[f0] + p1 * v[f0 + 1] + p2 * v[f0 + 2] + p3 * v[f0 + 3];
    for (int o = 16; o > 0; o >>= 1) s += __shfl_down(s, o);
    if (lane == 0) a_edge[e] = s;
    if (doMean) {
      atomicAdd(&smean[f0], p0 * invE);
      atomicAdd(&smean[f0 + 1], p1 * invE);
      atomicAdd(&smean[f0 + 2], p2 * invE);
      atomicAdd(&smean[f0 + 3], p3 * invE);
    }
  }
  if (doMean) {
    __syncthreads();
    for (int i = tid; i < 128; i += blockDim.x) atomicAdd(&meanAcc[i], smean[i]);
  }
}

// pass 1: alpha = lrelu0.2(as[s]+ad[d]+ae); store; segment-max into amax[d]
__global__ void k_alpha_max(const int* __restrict__ src, const int* __restrict__ dst,
                            const float* __restrict__ as_, const float* __restrict__ ad_,
                            const float* __restrict__ ae_, const float* __restrict__ loopA,
                            float* __restrict__ alpha_buf, float* __restrict__ amax,
                            int E, int ET) {
  int e = blockIdx.x * blockDim.x + threadIdx.x;
  if (e >= ET) return;
  int s, d; float aE;
  if (e < E) { s = src[e]; d = dst[e]; aE = ae_[e]; }
  else       { s = d = e - E; aE = loopA[0]; }
  float a = as_[s] + ad_[d] + aE;
  a = a > 0.0f ? a : 0.2f * a;
  alpha_buf[e] = a;
  atomicMaxF(&amax[d], a);
}

// pass 2: ex = exp(alpha - amax[d]); store; segment-sum into denom[d]
__global__ void k_exp_sum(const int* __restrict__ dst,
                          const float* __restrict__ alpha_buf,
                          const float* __restrict__ amax,
                          float* __restrict__ ex_buf, float* __restrict__ denom,
                          int E, int ET) {
  int e = blockIdx.x * blockDim.x + threadIdx.x;
  if (e >= ET) return;
  int d = (e < E) ? dst[e] : (e - E);
  float ex = __expf(alpha_buf[e] - amax[d]);
  ex_buf[e] = ex;
  atomicAdd(&denom[d], ex);
}

// pass 3: out[d] += hs[s] * ex/denom[d]   (wave per edge, float4 per lane)
__global__ void k_scatter(const int* __restrict__ src, const int* __restrict__ dst,
                          const float* __restrict__ ex_buf, const float* __restrict__ denom,
                          const float* __restrict__ hs, float* __restrict__ out,
                          int E, int ET) {
  int gid = blockIdx.x * blockDim.x + threadIdx.x;
  int e = gid >> 5, lane = gid & 31;
  if (e >= ET) return;
  int s, d;
  if (e < E) { s = src[e]; d = dst[e]; }
  else       { s = d = e - E; }
  float w = ex_buf[e] / (denom[d] + 1e-16f);
  float4 h = ((const float4*)(hs + (size_t)s * kH))[lane];
  float* o = out + (size_t)d * kH + lane * 4;
  atomicAdd(o + 0, h.x * w);
  atomicAdd(o + 1, h.y * w);
  atomicAdd(o + 2, h.z * w);
  atomicAdd(o + 3, h.w * w);
}

// out[r] = X[r,:] . wout + bout
__global__ void k_final(const float* __restrict__ X, const float* __restrict__ wout,
                        const float* __restrict__ bout, float* __restrict__ out, int M) {
  int gid = blockIdx.x * blockDim.x + threadIdx.x;
  int row = gid >> 5, lane = gid & 31;
  if (row >= M) return;
  float4 a = ((const float4*)(X + (size_t)row * kH))[lane];
  float4 b = ((const float4*)wout)[lane];
  float s = a.x * b.x + a.y * b.y + a.z * b.z + a.w * b.w;
  for (int o = 16; o > 0; o >>= 1) s += __shfl_down(s, o);
  if (lane == 0) out[row] = s + bout[0];
}

// ---------------------------------------------------------------------------
struct ConvScratch {
  float *hs, *a_src, *a_dst, *a_edge, *alpha, *ex, *amax, *denom;
  float *meanv, *wvec, *wvec2, *loopA;
};

// One GATConv relation. Wd==nullptr => same-type (alpha_dst from hs directly).
static void run_conv(hipStream_t st,
                     const float* Xsrc, int Ns, const float* Xdst, int Nd,
                     const int* src, const int* dst, int E,
                     const float* ea_raw, int De, const float* We_in, const float* be_in,
                     const float* Ws, const float* Wd,
                     const float* a_s, const float* a_d,
                     const float* Wedge, const float* ae,
                     bool loops, float* out, const ConvScratch& S) {
  // hs = Xsrc @ Ws  (WMMA)
  k_gemm_wmma<128><<<Ns / 16, 256, 0, st>>>(Xsrc, Ws, nullptr, S.hs, 1.0f);
  // alpha_src = hs . a_s
  k_rowdot<<<cdiv((long)Ns * 32, 256), 256, 0, st>>>(S.hs, a_s, S.a_src, Ns);
  // alpha_dst
  if (Wd == nullptr) {
    k_rowdot<<<cdiv((long)Ns * 32, 256), 256, 0, st>>>(S.hs, a_d, S.a_dst, Nd);
  } else {
    k_matvec128<<<1, 128, 0, st>>>(Wd, a_d, S.wvec2);
    k_rowdot<<<cdiv((long)Nd * 32, 256), 256, 0, st>>>(Xdst, S.wvec2, S.a_dst, Nd);
  }
  // wvec = Wedge @ ae ; alpha_e[e] = lrelu(ea@We_in+be_in) . wvec (fused)
  k_matvec128<<<1, 128, 0, st>>>(Wedge, ae, S.wvec);
  if (loops) k_fill<<<1, 128, 0, st>>>(S.meanv, 0.0f, 128);
  k_edge_alpha<<<cdiv((long)E * 32, 256), 256, 0, st>>>(
      ea_raw, De, We_in, be_in, S.wvec, S.a_edge, E,
      loops ? S.meanv : nullptr, 1.0f / (float)E);
  if (loops) k_dot128<<<1, 32, 0, st>>>(S.meanv, S.wvec, S.loopA);
  // segment softmax
  k_fill<<<cdiv(Nd, 256), 256, 0, st>>>(S.amax, -INFINITY, Nd);
  k_fill<<<cdiv(Nd, 256), 256, 0, st>>>(S.denom, 0.0f, Nd);
  int ET = E + (loops ? Nd : 0);
  k_alpha_max<<<cdiv(ET, 256), 256, 0, st>>>(src, dst, S.a_src, S.a_dst, S.a_edge,
                                             loops ? S.loopA : nullptr,
                                             S.alpha, S.amax, E, ET);
  k_exp_sum<<<cdiv(ET, 256), 256, 0, st>>>(dst, S.alpha, S.amax, S.ex, S.denom, E, ET);
  k_scatter<<<cdiv((long)ET * 32, 256), 256, 0, st>>>(src, dst, S.ex, S.denom,
                                                      S.hs, out, E, ET);
}

extern "C" void kernel_launch(void* const* d_in, const int* in_sizes, int n_in,
                              void* d_out, int out_size, void* d_ws, size_t ws_size,
                              hipStream_t stream) {
  (void)in_sizes; (void)n_in; (void)out_size; (void)ws_size;
  const float* x_prop   = (const float*)d_in[0];
  const float* x_branch = (const float*)d_in[1];
  const float* ea_pp = (const float*)d_in[2];
  const float* ea_bp = (const float*)d_in[3];
  const float* ea_bb = (const float*)d_in[4];
  const int* ei_pp = (const int*)d_in[5];
  const int* ei_bp = (const int*)d_in[6];
  const int* ei_bb = (const int*)d_in[7];
  const float* Wn_p = (const float*)d_in[8];   const float* bn_p = (const float*)d_in[9];
  const float* Wn_b = (const float*)d_in[10];  const float* bn_b = (const float*)d_in[11];
  const float* We_pp = (const float*)d_in[12]; const float* be_pp = (const float*)d_in[13];
  const float* We_bp = (const float*)d_in[14]; const float* be_bp = (const float*)d_in[15];
  const float* We_bb = (const float*)d_in[16]; const float* be_bb = (const float*)d_in[17];
  const float* W1_pp = (const float*)d_in[18];
  const float* as1_pp = (const float*)d_in[19]; const float* ad1_pp = (const float*)d_in[20];
  const float* Wedge1_pp = (const float*)d_in[21]; const float* ae1_pp = (const float*)d_in[22];
  const float* b1_pp = (const float*)d_in[23];
  const float* W1_bb = (const float*)d_in[24];
  const float* as1_bb = (const float*)d_in[25]; const float* ad1_bb = (const float*)d_in[26];
  const float* Wedge1_bb = (const float*)d_in[27]; const float* ae1_bb = (const float*)d_in[28];
  const float* b1_bb = (const float*)d_in[29];
  const float* Ws1_bp = (const float*)d_in[30]; const float* Wd1_bp = (const float*)d_in[31];
  const float* as1_bp = (const float*)d_in[32]; const float* ad1_bp = (const float*)d_in[33];
  const float* Wedge1_bp = (const float*)d_in[34]; const float* ae1_bp = (const float*)d_in[35];
  const float* b1_bp = (const float*)d_in[36];
  const float* W2_pp = (const float*)d_in[37];
  const float* as2_pp = (const float*)d_in[38]; const float* ad2_pp = (const float*)d_in[39];
  const float* Wedge2_pp = (const float*)d_in[40]; const float* ae2_pp = (const float*)d_in[41];
  const float* b2_pp = (const float*)d_in[42];
  // d_in[43..48] = *2_bb params, unused by reference conv2
  const float* Ws2_bp = (const float*)d_in[49]; const float* Wd2_bp = (const float*)d_in[50];
  const float* as2_bp = (const float*)d_in[51]; const float* ad2_bp = (const float*)d_in[52];
  const float* Wedge2_bp = (const float*)d_in[53]; const float* ae2_bp = (const float*)d_in[54];
  const float* b2_bp = (const float*)d_in[55];
  const float* Wout = (const float*)d_in[56];
  const float* bout = (const float*)d_in[57];

  // workspace carve (256B aligned chunks)
  char* w = (char*)d_ws;
  auto take = [&](size_t bytes) -> float* {
    char* p = w;
    w += (bytes + 255) & ~(size_t)255;
    return (float*)p;
  };
  float* xp  = take((size_t)kNPROP * kH * 4);  // projected prop feats / conv2 out
  float* xb  = take((size_t)kNBR   * kH * 4);
  float* xp2 = take((size_t)kNPROP * kH * 4);  // conv1 prop out
  float* xb2 = take((size_t)kNBR   * kH * 4);  // conv1 branch out
  ConvScratch S;
  S.hs     = take((size_t)kNBR * kH * 4);
  S.a_src  = take((size_t)kNBR * 4);
  S.a_dst  = take((size_t)kNBR * 4);
  S.a_edge = take((size_t)kEBB * 4);
  S.alpha  = take((size_t)(kEBB + kNBR) * 4);
  S.ex     = take((size_t)(kEBB + kNBR) * 4);
  S.amax   = take((size_t)kNBR * 4);
  S.denom  = take((size_t)kNBR * 4);
  S.meanv  = take(128 * 4);
  S.wvec   = take(128 * 4);
  S.wvec2  = take(128 * 4);
  S.loopA  = take(4);

  // 0) input projections (WMMA GEMMs, K=64, lrelu 0.01)
  k_gemm_wmma<64><<<kNPROP / 16, 256, 0, stream>>>(x_prop, Wn_p, bn_p, xp, 0.01f);
  k_gemm_wmma<64><<<kNBR / 16, 256, 0, stream>>>(x_branch, Wn_b, bn_b, xb, 0.01f);

  // 1) conv1 — HeteroConv sum aggregation: pre-init dst accumulators with biases
  k_init2<<<cdiv((long)kNPROP * kH, 256), 256, 0, stream>>>(xp2, b1_pp, b1_bp, kNPROP * kH);
  k_init2<<<cdiv((long)kNBR * kH, 256), 256, 0, stream>>>(xb2, b1_bb, nullptr, kNBR * kH);
  run_conv(stream, xp, kNPROP, xp, kNPROP, ei_pp, ei_pp + kEPP, kEPP,
           ea_pp, kDE, We_pp, be_pp, W1_pp, nullptr, as1_pp, ad1_pp,
           Wedge1_pp, ae1_pp, true, xp2, S);
  run_conv(stream, xb, kNBR, xp, kNPROP, ei_bp, ei_bp + kEBP, kEBP,
           ea_bp, kDE, We_bp, be_bp, Ws1_bp, Wd1_bp, as1_bp, ad1_bp,
           Wedge1_bp, ae1_bp, false, xp2, S);
  run_conv(stream, xb, kNBR, xb, kNBR, ei_bb, ei_bb + kEBB, kEBB,
           ea_bb, kDE, We_bb, be_bb, W1_bb, nullptr, as1_bb, ad1_bb,
           Wedge1_bb, ae1_bb, true, xb2, S);

  // 2) conv2 (inputs xp2/xb2; output reuses xp buffer)
  float* xp3 = xp;
  k_init2<<<cdiv((long)kNPROP * kH, 256), 256, 0, stream>>>(xp3, b2_pp, b2_bp, kNPROP * kH);
  run_conv(stream, xp2, kNPROP, xp2, kNPROP, ei_pp, ei_pp + kEPP, kEPP,
           ea_pp, kDE, We_pp, be_pp, W2_pp, nullptr, as2_pp, ad2_pp,
           Wedge2_pp, ae2_pp, true, xp3, S);
  run_conv(stream, xb2, kNBR, xp2, kNPROP, ei_bp, ei_bp + kEBP, kEBP,
           ea_bp, kDE, We_bp, be_bp, Ws2_bp, Wd2_bp, as2_bp, ad2_bp,
           Wedge2_bp, ae2_bp, false, xp3, S);

  // 3) readout
  k_final<<<cdiv((long)kNPROP * 32, 256), 256, 0, stream>>>(xp3, Wout, bout,
                                                            (float*)d_out, kNPROP);
}
